// MTRNNCell_50826642981639
// MI455X (gfx1250) — matrix operand order repaired and checked by reading, not compiled, not run on hardware
//
#include <hip/hip_runtime.h>
#include <math.h>

// ---------------- problem constants (from reference) ----------------
#define BTOT 8192
#define DE_  1024
#define DP_  64
#define DX_  64
#define FD_  160
#define SD_  64
#define UD_  32

typedef __attribute__((ext_vector_type(16))) _Float16     v16h;
typedef __attribute__((ext_vector_type(8)))  float        v8f;
typedef __attribute__((ext_vector_type(4)))  unsigned int v4u;
typedef __attribute__((ext_vector_type(8)))  int          v8i;
typedef __attribute__((ext_vector_type(4)))  int          v4i;

// ---- Tensor Data Mover availability (device pass only) ----
#if defined(__AMDGCN__) && defined(__has_builtin)
#  if __has_builtin(__builtin_amdgcn_tensor_load_to_lds) && __has_builtin(__builtin_amdgcn_s_wait_tensorcnt)
#    define HAVE_TDM 1
#  endif
#endif
#ifndef HAVE_TDM
#  define HAVE_TDM 0
#endif

// ---------------- weight handle: raw fp32 + optional f16 fragment-packed ----------------
struct WMat {
    const float*    raw;  // (K,N) row-major fp32
    const _Float16* pk;   // fragment-major f16: [(kt*ntiles+nt)*32 + lane]*16
};

// ---------------- A-fragment loader (16x32 f16, contiguous k order) ----------------
// lane L: m=L%16, khalf=L/16; elements j=0..15 -> k=j+16*kh
template <bool SCALED>
__device__ __forceinline__ v16h load_a(const float* __restrict__ X, int ldx,
                                       int rowBase, int kb, int lane, float s) {
    const int m  = lane & 15;
    const int kh = lane >> 4;
    const float* base = X + (size_t)(rowBase + m) * ldx + kb + 16 * kh;
    float4 c0 = *(const float4*)(base + 0);
    float4 c1 = *(const float4*)(base + 4);
    float4 c2 = *(const float4*)(base + 8);
    float4 c3 = *(const float4*)(base + 12);
    if (SCALED) {
        c0.x *= s; c0.y *= s; c0.z *= s; c0.w *= s;
        c1.x *= s; c1.y *= s; c1.z *= s; c1.w *= s;
        c2.x *= s; c2.y *= s; c2.z *= s; c2.w *= s;
        c3.x *= s; c3.y *= s; c3.z *= s; c3.w *= s;
    }
    v16h a;
    a[0]  = (_Float16)c0.x; a[1]  = (_Float16)c0.y; a[2]  = (_Float16)c0.z; a[3]  = (_Float16)c0.w;
    a[4]  = (_Float16)c1.x; a[5]  = (_Float16)c1.y; a[6]  = (_Float16)c1.z; a[7]  = (_Float16)c1.w;
    a[8]  = (_Float16)c2.x; a[9]  = (_Float16)c2.y; a[10] = (_Float16)c2.z; a[11] = (_Float16)c2.w;
    a[12] = (_Float16)c3.x; a[13] = (_Float16)c3.y; a[14] = (_Float16)c3.z; a[15] = (_Float16)c3.w;
    return a;
}

// ---------------- B-fragment loader ----------------
template <bool PACKED, int NT>
__device__ __forceinline__ v16h load_b(const WMat& W, int ldw, int kt, int nt, int lane) {
    if constexpr (PACKED) {
        const _Float16* p = W.pk + (((size_t)kt * NT + nt) * 32 + lane) * 16;
        return *(const v16h*)p;   // 32B contiguous -> 2x global_load_b128
    } else {
        const int n  = lane & 15;
        const int kh = lane >> 4;
        const float* wp = W.raw + (size_t)(kt * 32 + 16 * kh) * ldw + nt * 16 + n;
        v16h b;
#pragma unroll
        for (int v = 0; v < 8; ++v) {
            b[2 * v]     = (_Float16)wp[(2 * v) * ldw];
            b[2 * v + 1] = (_Float16)wp[(2 * v + 1) * ldw];
        }
        return b;
    }
}

// ---------------- streamed-A GEMM, software-pipelined one fragment ahead ----------------
template <bool PACKED, bool SCALED, int NT>
__device__ __forceinline__ void gemm(v8f* acc, const float* __restrict__ X, int ldx,
                                     const WMat& W, int ldw, int K, int rowBase, int lane,
                                     float scale) {
    const int KT = K / 32;
    v16h a = load_a<SCALED>(X, ldx, rowBase, 0, lane, scale);
    for (int kt = 0; kt < KT; ++kt) {
        // prefetch next A fragment (clamped index -> unconditional, branch-free)
        const int ktn = (kt + 1 < KT) ? kt + 1 : kt;
        v16h an = load_a<SCALED>(X, ldx, rowBase, ktn * 32, lane, scale);
        // B fragments pipelined one ahead so each load overlaps the previous WMMA
        v16h b = load_b<PACKED, NT>(W, ldw, kt, 0, lane);
#pragma unroll
        for (int nt = 0; nt < NT; ++nt) {
            const int ntn = (nt + 1 < NT) ? nt + 1 : nt;
            v16h bn = load_b<PACKED, NT>(W, ldw, kt, ntn, lane);
            acc[nt] = __builtin_amdgcn_wmma_f32_16x16x32_f16(
                false, a, false, b, (short)0, acc[nt], false, false);
            b = bn;
        }
        a = an;
    }
}

// ---------------- TDM staging for the streamed x_enc operand ----------------
#if HAVE_TDM
// tensor_load_to_lds of one 16x32 fp32 tile (row stride DE_) into LDS.
// D# per ISA 8.3/8.4: group0 = {count=1 | lds_addr | global_addr(57b) | type=2}
// group1 = {wg_mask=0,data_size=4B | dim0=1024 | dim1=8192 | tile0=32,tile1=16 | stride0=1024}
__device__ __forceinline__ void tdm_issue(const float* gsrc, unsigned ldsOff) {
    const unsigned long long ga = (unsigned long long)(size_t)gsrc;
    v4u g0 = { 1u, ldsOff, (unsigned)ga,
               ((unsigned)(ga >> 32) & 0x01FFFFFFu) | (2u << 30) };
    v8i g1 = { (int)0x00020000, (int)(DE_ << 16), (int)(BTOT << 16),
               (int)(32 << 16), 16, DE_, 0, 0 };
    v4i z4 = { 0, 0, 0, 0 };
#if __clang_major__ >= 23
    v8i z8 = { 0, 0, 0, 0, 0, 0, 0, 0 };
    __builtin_amdgcn_tensor_load_to_lds(g0, g1, z4, z4, z8, 0);
#else
    __builtin_amdgcn_tensor_load_to_lds(g0, g1, z4, z4, 0);
#endif
}
#endif

// x_enc GEMM: Tensor Data Mover keeps a 4-deep tile pipeline in LDS
template <bool PACKED, int NT>
__device__ __forceinline__ void gemm_tdm(v8f* acc, const float* __restrict__ xe,
                                         const WMat& W, int rowBase, int lane,
                                         float* buf, unsigned bufOff) {
#if HAVE_TDM
    const int KT = DE_ / 32;  // 32 K-tiles, 4 buffers
    const float* src = xe + (size_t)rowBase * DE_;
    asm volatile("s_wait_dscnt 0" ::: "memory");
    tdm_issue(src + 0 * 32, bufOff + 0 * 2048u);
    tdm_issue(src + 1 * 32, bufOff + 1 * 2048u);
    tdm_issue(src + 2 * 32, bufOff + 2 * 2048u);
    for (int kt = 0; kt < KT; ++kt) {
        const int cur = kt & 3;
        if (kt + 3 < KT) {
            asm volatile("s_wait_dscnt 0" ::: "memory");
            tdm_issue(src + (kt + 3) * 32, bufOff + (unsigned)(((kt + 3) & 3) * 2048));
            __builtin_amdgcn_s_wait_tensorcnt((short)3);
        } else if (kt + 2 < KT) {
            __builtin_amdgcn_s_wait_tensorcnt((short)2);
        } else if (kt + 1 < KT) {
            __builtin_amdgcn_s_wait_tensorcnt((short)1);
        } else {
            __builtin_amdgcn_s_wait_tensorcnt((short)0);
        }
        asm volatile("" ::: "memory");
        v16h a = load_a<false>(buf + cur * 512, 32, 0, 0, lane, 1.0f);  // ds_load from staged tile
        v16h b = load_b<PACKED, NT>(W, FD_, kt, 0, lane);
#pragma unroll
        for (int nt = 0; nt < NT; ++nt) {
            const int ntn = (nt + 1 < NT) ? nt + 1 : nt;
            v16h bn = load_b<PACKED, NT>(W, FD_, kt, ntn, lane);
            acc[nt] = __builtin_amdgcn_wmma_f32_16x16x32_f16(
                false, a, false, b, (short)0, acc[nt], false, false);
            b = bn;
        }
        asm volatile("" ::: "memory");
    }
#else
    (void)buf; (void)bufOff;
    gemm<PACKED, false, NT>(acc, xe, DE_, W, FD_, DE_, rowBase, lane, 1.0f);
#endif
}

// ---------------- C/D store: leaky blend + tanh ----------------
template <int NT>
__device__ __forceinline__ void store_out(const v8f* acc, const float* __restrict__ prev,
                                          float alpha, float* __restrict__ outTanh,
                                          float* __restrict__ outRaw, int ld,
                                          int rowBase, int lane) {
    const int n  = lane & 15;
    const int mh = lane >> 4;
#pragma unroll
    for (int nt = 0; nt < NT; ++nt) {
#pragma unroll
        for (int r = 0; r < 8; ++r) {
            const int row    = rowBase + r + 8 * mh;
            const size_t idx = (size_t)row * ld + nt * 16 + n;
            const float raw  = (1.0f - alpha) * prev[idx] + alpha * acc[nt][r];
            outRaw[idx]  = raw;
            outTanh[idx] = tanhf(raw);
        }
    }
}

__device__ __forceinline__ int strip_base() {
    return (blockIdx.x * (blockDim.x >> 5) + (threadIdx.x >> 5)) * 16;
}

// ================= kernels: one per output family (small kernargs, low VGPR) =================

// new_u{p,x} = (1-af)*u + af*(x@Wi2f + h@Wf2f + hs@Ws2f); outputs tanh + raw
template <bool PACKED>
__global__ __launch_bounds__(128) void k_fast(const float* __restrict__ x, int Kx, WMat Wi2f,
                                              const float* __restrict__ h,  WMat Wf2f,
                                              const float* __restrict__ hs, WMat Ws2f,
                                              const float* __restrict__ uprev,
                                              float* __restrict__ tOut, float* __restrict__ rOut) {
    const int lane = threadIdx.x & 31;
    const int rowBase = strip_base();
    v8f acc[10] = {};
    gemm<PACKED, false, 10>(acc, x,  Kx,  Wi2f, FD_, Kx,  rowBase, lane, 1.0f);
    gemm<PACKED, false, 10>(acc, h,  FD_, Wf2f, FD_, FD_, rowBase, lane, 1.0f);
    gemm<PACKED, false, 10>(acc, hs, SD_, Ws2f, FD_, SD_, rowBase, lane, 1.0f);
    store_out<10>(acc, uprev, 0.5f, tOut, rOut, FD_, rowBase, lane);
}

// new_ue : x_enc GEMM staged through LDS by the Tensor Data Mover
template <bool PACKED>
__global__ __launch_bounds__(128) void k_enc(const float* __restrict__ xe, WMat Wi2f,
                                             const float* __restrict__ he, WMat Wf2f,
                                             const float* __restrict__ hs, WMat Ws2f,
                                             const float* __restrict__ uprev,
                                             float* __restrict__ tOut, float* __restrict__ rOut) {
    __shared__ float s_xe[4][4][512];   // sole LDS alloc -> base offset 0 for D# lds_addr
    const int lane = threadIdx.x & 31;
    const int wave = threadIdx.x >> 5;
    const int rowBase = strip_base();
    v8f acc[10] = {};
    gemm_tdm<PACKED, 10>(acc, xe, Wi2f, rowBase, lane, &s_xe[wave][0][0], (unsigned)(wave * 8192));
    gemm<PACKED, false, 10>(acc, he, FD_, Wf2f, FD_, FD_, rowBase, lane, 1.0f);
    gemm<PACKED, false, 10>(acc, hs, SD_, Ws2f, FD_, SD_, rowBase, lane, 1.0f);
    store_out<10>(acc, uprev, 0.5f, tOut, rOut, FD_, rowBase, lane);
}

// new_us : hp@pose_f2s + (he*g1)@enc_f2s + (hx*g3)@prox_f2s + hs@s2s + h_un@u2s
template <bool PACKED>
__global__ __launch_bounds__(128) void k_slow(const float* __restrict__ hp, WMat Wpf2s,
                                              const float* __restrict__ he, WMat Wef2s,
                                              const float* __restrict__ hx, WMat Wxf2s,
                                              const float* __restrict__ hs, WMat Ws2s,
                                              const float* __restrict__ hun, WMat Wu2s,
                                              const float* __restrict__ katt,
                                              const float* __restrict__ batt,
                                              const float* __restrict__ uprev,
                                              float* __restrict__ tOut, float* __restrict__ rOut) {
    const int lane = threadIdx.x & 31;
    const int rowBase = strip_base();

    // attention gate: each lane computes its row's normalized (a1,a3)
    float s1, s3;
    {
        const int row = rowBase + (lane & 15);
        float d0 = 0.f, d1 = 0.f;
        for (int k = 0; k < SD_; ++k) {
            const float h = hs[(size_t)row * SD_ + k];
            d0 += h * katt[k * 2 + 0];
            d1 += h * katt[k * 2 + 1];
        }
        const float a1  = 1.0f / (1.0f + __expf(-(d0 + batt[0])));
        const float a3  = 1.0f / (1.0f + __expf(-(d1 + batt[1])));
        const float den = a1 + a3 + 1e-7f;
        s1 = a1 / den;
        s3 = a3 / den;
    }

    v8f acc[4] = {};
    gemm<PACKED, false, 4>(acc, hp,  FD_, Wpf2s, SD_, FD_, rowBase, lane, 1.0f);
    gemm<PACKED, true,  4>(acc, he,  FD_, Wef2s, SD_, FD_, rowBase, lane, s1);
    gemm<PACKED, true,  4>(acc, hx,  FD_, Wxf2s, SD_, FD_, rowBase, lane, s3);
    gemm<PACKED, false, 4>(acc, hs,  SD_, Ws2s,  SD_, SD_, rowBase, lane, 1.0f);
    gemm<PACKED, false, 4>(acc, hun, UD_, Wu2s,  SD_, UD_, rowBase, lane, 1.0f);
    store_out<4>(acc, uprev, 1.0f / 12.0f, tOut, rOut, SD_, rowBase, lane);
}

// union : new_u_un = (1-au)*u_un + au*(l_hs@Ls2u + r_hs@Rs2u + h_un@u2u)
template <bool PACKED>
__global__ __launch_bounds__(128) void k_union(const float* __restrict__ hsL, WMat WLs2u,
                                               const float* __restrict__ hsR, WMat WRs2u,
                                               const float* __restrict__ hun, WMat Wu2u,
                                               const float* __restrict__ uprev,
                                               float* __restrict__ tOut, float* __restrict__ rOut) {
    const int lane = threadIdx.x & 31;
    const int rowBase = strip_base();
    v8f acc[2] = {};
    gemm<PACKED, false, 2>(acc, hsL, SD_, WLs2u, UD_, SD_, rowBase, lane, 1.0f);
    gemm<PACKED, false, 2>(acc, hsR, SD_, WRs2u, UD_, SD_, rowBase, lane, 1.0f);
    gemm<PACKED, false, 2>(acc, hun, UD_, Wu2u,  UD_, UD_, rowBase, lane, 1.0f);
    store_out<2>(acc, uprev, 1.0f / 25.0f, tOut, rOut, UD_, rowBase, lane);
}

// ---------------- weight prep: fp32 (K,N) -> f16 fragment-major ----------------
__global__ void pack_weights(const float* __restrict__ W, _Float16* __restrict__ dst,
                             int K, int N) {
    const int ntiles = N / 16;
    const int kt   = blockIdx.x / ntiles;
    const int nt   = blockIdx.x % ntiles;
    const int lane = threadIdx.x;           // 32 threads
    const int n    = lane & 15;
    const int kh   = lane >> 4;
    _Float16* out   = dst + (((size_t)kt * ntiles + nt) * 32 + lane) * 16;
    const float* in = W + (size_t)(kt * 32 + 16 * kh) * N + nt * 16 + n;
#pragma unroll
    for (int j = 0; j < 16; ++j) out[j] = (_Float16)in[j * N];
}

// ---------------- host-side launch ----------------
extern "C" void kernel_launch(void* const* d_in, const int* in_sizes, int n_in,
                              void* d_out, int out_size, void* d_ws, size_t ws_size,
                              hipStream_t stream) {
    (void)n_in; (void)out_size;

    const int LBASE = 24, RBASE = 39;
    // params pytree flattened with sorted dict keys (per-side order below)
    struct MD { int K, N; };
    static const MD mds[15] = {
        {160, 160}, {160, 64}, {1024, 160}, {64, 160},   // enc_f2f, enc_f2s, enc_i2f, enc_s2f
        {160, 160}, {160, 64}, {64, 160},   {64, 160},   // pose_f2f, pose_f2s, pose_i2f, pose_s2f
        {160, 160}, {160, 64}, {64, 160},   {64, 160},   // prox_f2f, prox_f2s, prox_i2f, prox_s2f
        {64, 64},   {64, 32},  {32, 64}                  // s2s, s2u, u2s
    };

    size_t needElems = 0;
    for (int i = 0; i < 15; ++i) needElems += (size_t)mds[i].K * mds[i].N;
    needElems = needElems * 2 + (size_t)UD_ * UD_;
    const bool packed = (ws_size >= needElems * sizeof(_Float16));

    _Float16* pkp[31];
    if (packed) {
        _Float16* wsf = (_Float16*)d_ws;
        size_t off = 0;
        for (int s = 0; s < 2; ++s) {
            for (int i = 0; i < 15; ++i) {
                const int K = mds[i].K, N = mds[i].N;
                const float* src = (const float*)d_in[(s ? RBASE : LBASE) + i];
                pkp[s * 15 + i] = wsf + off;
                pack_weights<<<(K / 32) * (N / 16), 32, 0, stream>>>(src, wsf + off, K, N);
                off += (size_t)K * N;
            }
        }
        pkp[30] = wsf + off;
        pack_weights<<<(UD_ / 32) * (UD_ / 16), 32, 0, stream>>>(
            (const float*)d_in[56], wsf + off, UD_, UD_);
    } else {
        for (int i = 0; i < 31; ++i) pkp[i] = nullptr;
    }

    // sorted per-side weight indices
    enum { W_ENC_F2F = 0, W_ENC_F2S, W_ENC_I2F, W_ENC_S2F,
           W_POSE_F2F, W_POSE_F2S, W_POSE_I2F, W_POSE_S2F,
           W_PROX_F2F, W_PROX_F2S, W_PROX_I2F, W_PROX_S2F,
           W_S2S, W_S2U, W_U2S };
    auto wm = [&](int side, int wi) -> WMat {
        return { (const float*)d_in[(side ? RBASE : LBASE) + wi], pkp[side * 15 + wi] };
    };
    WMat u2u = { (const float*)d_in[56], pkp[30] };
    const float* batt = (const float*)d_in[54];
    const float* katt = (const float*)d_in[55];
    const float* h_un = (const float*)d_in[22];
    const float* u_un = (const float*)d_in[23];

    // output layout: (t_up,t_ue,t_ux,t_us, up,ue,ux,us) x {L,R}, then h_un, u_un
    const int B = in_sizes[0] / DP_;
    const size_t SZF = (size_t)B * FD_;
    const size_t SZS = (size_t)B * SD_;
    const size_t SZU = (size_t)B * UD_;
    float* o = (float*)d_out;
    const size_t sideStride = 6 * SZF + 2 * SZS;

    const int grid = B / 64;  // 4 waves x 16 rows per block
    const int blk  = 128;

    for (int s = 0; s < 2; ++s) {
        const int ib = s ? 3 : 0;    // xr_* vs xl_*
        const int hb = s ? 14 : 6;   // r*_hf vs l*_hf
        const int ub = s ? 18 : 10;  // r*_uf vs l*_uf
        const float* xp = (const float*)d_in[ib + 0];
        const float* xe = (const float*)d_in[ib + 1];
        const float* xx = (const float*)d_in[ib + 2];
        const float* hp = (const float*)d_in[hb + 0];
        const float* he = (const float*)d_in[hb + 1];
        const float* hx = (const float*)d_in[hb + 2];
        const float* hs = (const float*)d_in[hb + 3];
        const float* up = (const float*)d_in[ub + 0];
        const float* ue = (const float*)d_in[ub + 1];
        const float* ux = (const float*)d_in[ub + 2];
        const float* us = (const float*)d_in[ub + 3];
        float* base = o + (size_t)s * sideStride;
        float* t_up = base;            float* t_ue = base + SZF;
        float* t_ux = base + 2 * SZF;  float* t_us = base + 3 * SZF;
        float* o_up = base + 3 * SZF + SZS;
        float* o_ue = base + 4 * SZF + SZS;
        float* o_ux = base + 5 * SZF + SZS;
        float* o_us = base + 6 * SZF + SZS;

        if (packed) {
            k_enc<true><<<grid, blk, 0, stream>>>(xe, wm(s, W_ENC_I2F), he, wm(s, W_ENC_F2F),
                                                  hs, wm(s, W_ENC_S2F), ue, t_ue, o_ue);
            k_fast<true><<<grid, blk, 0, stream>>>(xp, DP_, wm(s, W_POSE_I2F), hp, wm(s, W_POSE_F2F),
                                                   hs, wm(s, W_POSE_S2F), up, t_up, o_up);
            k_fast<true><<<grid, blk, 0, stream>>>(xx, DX_, wm(s, W_PROX_I2F), hx, wm(s, W_PROX_F2F),
                                                   hs, wm(s, W_PROX_S2F), ux, t_ux, o_ux);
            k_slow<true><<<grid, blk, 0, stream>>>(hp, wm(s, W_POSE_F2S), he, wm(s, W_ENC_F2S),
                                                   hx, wm(s, W_PROX_F2S), hs, wm(s, W_S2S),
                                                   h_un, wm(s, W_U2S), katt, batt, us, t_us, o_us);
        } else {
            k_enc<false><<<grid, blk, 0, stream>>>(xe, wm(s, W_ENC_I2F), he, wm(s, W_ENC_F2F),
                                                   hs, wm(s, W_ENC_S2F), ue, t_ue, o_ue);
            k_fast<false><<<grid, blk, 0, stream>>>(xp, DP_, wm(s, W_POSE_I2F), hp, wm(s, W_POSE_F2F),
                                                    hs, wm(s, W_POSE_S2F), up, t_up, o_up);
            k_fast<false><<<grid, blk, 0, stream>>>(xx, DX_, wm(s, W_PROX_I2F), hx, wm(s, W_PROX_F2F),
                                                    hs, wm(s, W_PROX_S2F), ux, t_ux, o_ux);
            k_slow<false><<<grid, blk, 0, stream>>>(hp, wm(s, W_POSE_F2S), he, wm(s, W_ENC_F2S),
                                                    hx, wm(s, W_PROX_F2S), hs, wm(s, W_S2S),
                                                    h_un, wm(s, W_U2S), katt, batt, us, t_us, o_us);
        }
    }

    const float* hsL = (const float*)d_in[9];
    const float* hsR = (const float*)d_in[17];
    float* o_hun = o + 2 * sideStride;
    float* o_uun = o + 2 * sideStride + SZU;
    if (packed)
        k_union<true><<<grid, blk, 0, stream>>>(hsL, wm(0, W_S2U), hsR, wm(1, W_S2U),
                                                h_un, u2u, u_un, o_hun, o_uun);
    else
        k_union<false><<<grid, blk, 0, stream>>>(hsL, wm(0, W_S2U), hsR, wm(1, W_S2U),
                                                 h_un, u2u, u_un, o_hun, o_uun);
}
